// TinyBlock_2070174237099
// MI455X (gfx1250) — matrix operand order repaired and checked by reading, not compiled, not run on hardware
//
#include <hip/hip_runtime.h>

// ---------------------------------------------------------------------------
// Types
// ---------------------------------------------------------------------------
typedef __bf16 bf16_t;
typedef __bf16 v16bf __attribute__((ext_vector_type(16)));
typedef float  v8f   __attribute__((ext_vector_type(8)));
typedef unsigned int v4u __attribute__((ext_vector_type(4)));

union Frag {              // one WMMA bf16 A/B operand (16 halves per lane)
  v16bf v;
  v4u   q[2];
};

__device__ __forceinline__ bf16_t f2bf(float f) {
  union { float f; unsigned int u; } a; a.f = f;
  unsigned int r = a.u + 0x7FFFu + ((a.u >> 16) & 1u);   // round-to-nearest-even
  union { unsigned short s; bf16_t b; } o; o.s = (unsigned short)(r >> 16);
  return o.b;
}

// CDNA5 async copy: 16 bytes / lane, global -> LDS, tracked by ASYNCcnt.
__device__ __forceinline__ void async_ld16(unsigned lds_off, const bf16_t* g) {
  asm volatile("global_load_async_to_lds_b128 %0, %1, off"
               :: "v"(lds_off), "v"(g) : "memory");
}

// Problem constants
#define MT   4096   // B*T rows
#define CC   1024   // n_embd
#define FF   4096   // d_ff
#define TT   2048   // seq len
#define NH   16
#define DH   64

// ---------------------------------------------------------------------------
// fp32 -> bf16 convert (grid-stride)
// ---------------------------------------------------------------------------
__global__ void k_to_bf16(const float* __restrict__ in, bf16_t* __restrict__ out, int n) {
  for (int i = blockIdx.x * blockDim.x + threadIdx.x; i < n; i += gridDim.x * blockDim.x)
    out[i] = f2bf(in[i]);
}

// ---------------------------------------------------------------------------
// Generic WMMA GEMM:  out[m,n] = sum_k A[m,k] * W[n,k]  (+bias, opt. GELU)
// A: [M,K] bf16 row-major, W: [N,K] bf16 row-major.
// 128x128 tile / block (256 thr = 8 waves), BLK_K = 32.
// Triple-buffered LDS fed by GLOBAL_LOAD_ASYNC_TO_LDS_B128, 2 tiles ahead,
// one barrier per K-step.
// ---------------------------------------------------------------------------
#define BLK_M 128
#define BLK_N 128
#define BLK_K 32
#define APITCH 40          // halves per LDS row (32 data + 8 pad -> 20-bank stride)

__global__ __launch_bounds__(256)
void k_gemm_bf16(const bf16_t* __restrict__ A, const bf16_t* __restrict__ W,
                 const float* __restrict__ bias,
                 float* __restrict__ outF, bf16_t* __restrict__ outB,
                 int Ndim, int Kdim, int gelu)
{
  __shared__ __align__(16) bf16_t sA[3][BLK_M * APITCH];
  __shared__ __align__(16) bf16_t sW[3][BLK_N * APITCH];

  const int tid  = threadIdx.x;
  const int lane = tid & 31;
  const int w    = tid >> 5;       // wave 0..7
  const int wm   = w & 3;          // 4 waves along M (32 rows each)
  const int wn   = w >> 2;         // 2 waves along N (64 cols each)
  const int hi   = lane >> 4;      // lane half
  const int l16  = lane & 15;
  const int m0   = blockIdx.x * BLK_M;
  const int n0   = blockIdx.y * BLK_N;
  const int nk   = Kdim / BLK_K;   // always >= 2 here

  // issue one tile's async loads: per thread 2x16B for A and 2x16B for W
  // => 4 ASYNCcnt increments per wave per tile.
  auto issue = [&](int kt, int buf) {
    const bf16_t* Ab = A + (size_t)m0 * Kdim + kt * BLK_K;
    const bf16_t* Wb = W + (size_t)n0 * Kdim + kt * BLK_K;
#pragma unroll
    for (int i = 0; i < 2; ++i) {
      int idx = tid + i * 256;
      int rr = idx >> 2, cc = (idx & 3) * 8;
      unsigned la = (unsigned)(unsigned long long)(void*)&sA[buf][rr * APITCH + cc];
      unsigned lw = (unsigned)(unsigned long long)(void*)&sW[buf][rr * APITCH + cc];
      async_ld16(la, Ab + (size_t)rr * Kdim + cc);
      async_ld16(lw, Wb + (size_t)rr * Kdim + cc);
    }
  };

  v8f acc[2][4];
#pragma unroll
  for (int mi = 0; mi < 2; ++mi)
#pragma unroll
    for (int ni = 0; ni < 4; ++ni)
#pragma unroll
      for (int j = 0; j < 8; ++j) acc[mi][ni][j] = 0.f;

  issue(0, 0);
  issue(1, 1);

  for (int kt = 0; kt < nk; ++kt) {
    // tile kt's 4 async ops are the oldest; at most tile kt+1's 4 may remain.
    if (kt + 1 < nk) asm volatile("s_wait_asynccnt 0x4" ::: "memory");
    else             asm volatile("s_wait_asynccnt 0x0" ::: "memory");
    __syncthreads();                       // all waves' tile-kt data visible
    if (kt + 2 < nk) issue(kt + 2, (kt + 2) % 3);

    const bf16_t* bufA = sA[kt % 3];
    const bf16_t* bufW = sW[kt % 3];

    Frag a[2], b[4];
#pragma unroll
    for (int mi = 0; mi < 2; ++mi) {
      const bf16_t* base = &bufA[(wm * 32 + mi * 16 + l16) * APITCH];
      a[mi].q[0] = *(const v4u*)&base[8 * hi];        // K = 8*hi .. 8*hi+7
      a[mi].q[1] = *(const v4u*)&base[16 + 8 * hi];   // K = 16+8*hi ..
    }
#pragma unroll
    for (int ni = 0; ni < 4; ++ni) {
      const bf16_t* base = &bufW[(wn * 64 + ni * 16 + l16) * APITCH];
      b[ni].q[0] = *(const v4u*)&base[8 * hi];        // K = 8*hi .. 8*hi+7
      b[ni].q[1] = *(const v4u*)&base[16 + 8 * hi];
    }
#pragma unroll
    for (int mi = 0; mi < 2; ++mi)
#pragma unroll
      for (int ni = 0; ni < 4; ++ni)
        acc[mi][ni] = __builtin_amdgcn_wmma_f32_16x16x32_bf16(
            false, a[mi].v, false, b[ni].v, (short)0, acc[mi][ni], false, false);
  }

  // epilogue
#pragma unroll
  for (int mi = 0; mi < 2; ++mi)
#pragma unroll
    for (int ni = 0; ni < 4; ++ni) {
      int n = n0 + wn * 64 + ni * 16 + l16;
      float bv = bias ? bias[n] : 0.f;
#pragma unroll
      for (int j = 0; j < 8; ++j) {
        int m = m0 + wm * 32 + mi * 16 + j + 8 * hi;
        float val = acc[mi][ni][j] + bv;
        if (gelu) val = 0.5f * val * (1.f + erff(val * 0.70710678118f));
        size_t off = (size_t)m * Ndim + n;
        if (outF) outF[off] = val;
        if (outB) outB[off] = f2bf(val);
      }
    }
}

// ---------------------------------------------------------------------------
// Flash attention (causal). Block = 128 thr (4 waves), owns 64 q-rows of one
// (batch, head). Streams 64-key tiles with online softmax; all matmuls WMMA.
// q/k/v stored as [B*T, C] bf16 (head h at column h*64).
// ---------------------------------------------------------------------------
#define KPITCH 72          // halves per LDS row (64 + 8 pad -> 36-bank stride)

__global__ __launch_bounds__(128)
void k_attention(const bf16_t* __restrict__ q, const bf16_t* __restrict__ k,
                 const bf16_t* __restrict__ v, bf16_t* __restrict__ out)
{
  __shared__ __align__(16) bf16_t sK[64 * KPITCH];          // [key][d]
  __shared__ __align__(16) bf16_t sV[64 * KPITCH];          // transposed: [d][key]
  __shared__ __align__(16) bf16_t sP[4][16 * KPITCH];       // per-wave P tile [m][key]

  const int tid  = threadIdx.x;
  const int lane = tid & 31;
  const int w    = tid >> 5;
  const int hi   = lane >> 4;
  const int l16  = lane & 15;
  const int qt   = blockIdx.x;                // q tile (64 rows)
  const int b    = blockIdx.y >> 4;
  const int h    = blockIdx.y & 15;
  const int q0   = qt * 64;
  const int hoff = h * DH;

  // Q fragments (reused across all key tiles): rows q0 + w*16 + l16
  Frag qa[2];
  {
    const bf16_t* qb = q + (size_t)(b * TT + q0 + w * 16 + l16) * CC + hoff;
#pragma unroll
    for (int c = 0; c < 2; ++c) {
      qa[c].q[0] = *(const v4u*)&qb[c * 32 + 8 * hi];
      qa[c].q[1] = *(const v4u*)&qb[c * 32 + 16 + 8 * hi];
    }
  }

  float mstat[8], lstat[8];
  v8f o[4];
#pragma unroll
  for (int j = 0; j < 8; ++j) { mstat[j] = -1e30f; lstat[j] = 0.f; }
#pragma unroll
  for (int di = 0; di < 4; ++di)
#pragma unroll
    for (int j = 0; j < 8; ++j) o[di][j] = 0.f;

  for (int kt = 0; kt <= qt; ++kt) {
    __syncthreads();
    // load K tile, and V tile transposed
#pragma unroll
    for (int i = 0; i < 4; ++i) {
      int idx = tid + i * 128;
      int key = idx >> 3, c8 = (idx & 7) * 8;
      size_t grow = (size_t)(b * TT + kt * 64 + key) * CC + hoff + c8;
      *(v4u*)&sK[key * KPITCH + c8] = *(const v4u*)(k + grow);
      union { v4u u; bf16_t hh[8]; } t;
      t.u = *(const v4u*)(v + grow);
#pragma unroll
      for (int e = 0; e < 8; ++e) sV[(c8 + e) * KPITCH + key] = t.hh[e];
    }
    __syncthreads();

    // S = Q K^T  (4 key-frags x 2 d-chunks)
    v8f s[4];
#pragma unroll
    for (int ni = 0; ni < 4; ++ni) {
#pragma unroll
      for (int j = 0; j < 8; ++j) s[ni][j] = 0.f;
#pragma unroll
      for (int c = 0; c < 2; ++c) {
        Frag kb;
        const bf16_t* base = &sK[(ni * 16 + l16) * KPITCH + c * 32];
        kb.q[0] = *(const v4u*)&base[8 * hi];
        kb.q[1] = *(const v4u*)&base[16 + 8 * hi];
        s[ni] = __builtin_amdgcn_wmma_f32_16x16x32_bf16(
            false, qa[c].v, false, kb.v, (short)0, s[ni], false, false);
      }
    }

    // scale + causal mask (only needed on the diagonal tile)
    const bool diag = (kt == qt);
#pragma unroll
    for (int ni = 0; ni < 4; ++ni)
#pragma unroll
      for (int j = 0; j < 8; ++j) {
        float sv = s[ni][j] * 0.125f;                 // 1/sqrt(64)
        if (diag) {
          int key = kt * 64 + ni * 16 + l16;
          int qr  = q0 + w * 16 + j + 8 * hi;
          if (key > qr) sv = -1e30f;
        }
        s[ni][j] = sv;
      }

    // online softmax per row (row r = j + 8*hi lives in a 16-lane half)
#pragma unroll
    for (int j = 0; j < 8; ++j) {
      float mx = s[0][j];
#pragma unroll
      for (int ni = 1; ni < 4; ++ni) mx = fmaxf(mx, s[ni][j]);
#pragma unroll
      for (int d = 1; d < 16; d <<= 1) mx = fmaxf(mx, __shfl_xor(mx, d, 32));
      float mnew  = fmaxf(mstat[j], mx);
      float scale = __expf(mstat[j] - mnew);
      mstat[j] = mnew;
      float rs = 0.f;
#pragma unroll
      for (int ni = 0; ni < 4; ++ni) {
        float p = __expf(s[ni][j] - mnew);
        s[ni][j] = p;
        rs += p;
      }
#pragma unroll
      for (int d = 1; d < 16; d <<= 1) rs += __shfl_xor(rs, d, 32);
      lstat[j] = lstat[j] * scale + rs;
#pragma unroll
      for (int di = 0; di < 4; ++di) o[di][j] *= scale;
    }

    // spill P (bf16) to this wave's LDS region, re-read as A fragments
#pragma unroll
    for (int ni = 0; ni < 4; ++ni)
#pragma unroll
      for (int j = 0; j < 8; ++j)
        sP[w][(j + 8 * hi) * KPITCH + ni * 16 + l16] = f2bf(s[ni][j]);
    __syncthreads();

    Frag pa[2];
#pragma unroll
    for (int c = 0; c < 2; ++c) {
      const bf16_t* base = &sP[w][l16 * KPITCH + c * 32];
      pa[c].q[0] = *(const v4u*)&base[8 * hi];
      pa[c].q[1] = *(const v4u*)&base[16 + 8 * hi];
    }
#pragma unroll
    for (int di = 0; di < 4; ++di)
#pragma unroll
      for (int c = 0; c < 2; ++c) {
        Frag vb;
        const bf16_t* base = &sV[(di * 16 + l16) * KPITCH + c * 32];
        vb.q[0] = *(const v4u*)&base[8 * hi];
        vb.q[1] = *(const v4u*)&base[16 + 8 * hi];
        o[di] = __builtin_amdgcn_wmma_f32_16x16x32_bf16(
            false, pa[c].v, false, vb.v, (short)0, o[di], false, false);
      }
  }

  // normalize and write out (bf16, [B*T, C] layout)
  float rinv[8];
#pragma unroll
  for (int j = 0; j < 8; ++j) rinv[j] = 1.f / lstat[j];
#pragma unroll
  for (int di = 0; di < 4; ++di)
#pragma unroll
    for (int j = 0; j < 8; ++j) {
      int qr = q0 + w * 16 + j + 8 * hi;
      out[(size_t)(b * TT + qr) * CC + hoff + di * 16 + l16] = f2bf(o[di][j] * rinv[j]);
    }
}

// ---------------------------------------------------------------------------
// y = LayerNorm(a + r) * g + be ; one block per row of C=1024
// ---------------------------------------------------------------------------
__global__ __launch_bounds__(256)
void k_add_ln(const float* __restrict__ a, const float* __restrict__ r,
              const float* __restrict__ g, const float* __restrict__ be,
              float* __restrict__ yF, bf16_t* __restrict__ yB)
{
  __shared__ float red[16];
  const int row = blockIdx.x;
  const int tid = threadIdx.x;
  const int lane = tid & 31, w = tid >> 5;
  float x[4], sum = 0.f, sq = 0.f;
#pragma unroll
  for (int i = 0; i < 4; ++i) {
    int c = tid + i * 256;
    x[i] = a[(size_t)row * CC + c] + r[(size_t)row * CC + c];
    sum += x[i]; sq += x[i] * x[i];
  }
#pragma unroll
  for (int d = 1; d < 32; d <<= 1) { sum += __shfl_xor(sum, d, 32); sq += __shfl_xor(sq, d, 32); }
  if (lane == 0) { red[w] = sum; red[8 + w] = sq; }
  __syncthreads();
  float ts = 0.f, tq = 0.f;
#pragma unroll
  for (int i = 0; i < 8; ++i) { ts += red[i]; tq += red[8 + i]; }
  const float mu = ts * (1.f / CC);
  const float var = tq * (1.f / CC) - mu * mu;
  const float rstd = rsqrtf(var + 1e-5f);
#pragma unroll
  for (int i = 0; i < 4; ++i) {
    int c = tid + i * 256;
    float y = (x[i] - mu) * rstd * g[c] + be[c];
    if (yF) yF[(size_t)row * CC + c] = y;
    if (yB) yB[(size_t)row * CC + c] = f2bf(y);
  }
}

// ---------------------------------------------------------------------------
// Host-side orchestration
// ---------------------------------------------------------------------------
extern "C" void kernel_launch(void* const* d_in, const int* in_sizes, int n_in,
                              void* d_out, int out_size, void* d_ws, size_t ws_size,
                              hipStream_t stream)
{
  (void)in_sizes; (void)n_in; (void)out_size; (void)ws_size;
  const float* x     = (const float*)d_in[0];
  // d_in[1] = mask (causal; not needed)
  const float* wq_w  = (const float*)d_in[2];
  const float* wq_b  = (const float*)d_in[3];
  const float* wk_w  = (const float*)d_in[4];
  const float* wk_b  = (const float*)d_in[5];
  const float* wv_w  = (const float*)d_in[6];
  const float* wv_b  = (const float*)d_in[7];
  const float* wo_w  = (const float*)d_in[8];
  const float* wo_b  = (const float*)d_in[9];
  const float* ln1_g = (const float*)d_in[10];
  const float* ln1_b = (const float*)d_in[11];
  const float* ln2_g = (const float*)d_in[12];
  const float* ln2_b = (const float*)d_in[13];
  const float* ff1_w = (const float*)d_in[14];
  const float* ff1_b = (const float*)d_in[15];
  const float* ff2_w = (const float*)d_in[16];
  const float* ff2_b = (const float*)d_in[17];

  char* ws = (char*)d_ws;
  const size_t MB = 1024ull * 1024ull;
  bf16_t* wq_bf  = (bf16_t*)(ws + 0 * MB);    // 2 MB each
  bf16_t* wk_bf  = (bf16_t*)(ws + 2 * MB);
  bf16_t* wv_bf  = (bf16_t*)(ws + 4 * MB);
  bf16_t* wo_bf  = (bf16_t*)(ws + 6 * MB);
  bf16_t* ff1_bf = (bf16_t*)(ws + 8 * MB);    // 8 MB
  bf16_t* ff2_bf = (bf16_t*)(ws + 16 * MB);   // 8 MB
  bf16_t* q_bf   = (bf16_t*)(ws + 24 * MB);   // 8 MB
  bf16_t* k_bf   = (bf16_t*)(ws + 32 * MB);   // 8 MB
  bf16_t* v_bf   = (bf16_t*)(ws + 40 * MB);   // 8 MB
  bf16_t* x_bf   = (bf16_t*)(ws + 48 * MB);   // 8 MB
  bf16_t* att_bf = (bf16_t*)(ws + 48 * MB);   // reuse x_bf (dead after QKV)
  float*  proj_f = (float*) (ws + 56 * MB);   // 16 MB
  float*  x1_f   = (float*) (ws + 72 * MB);   // 16 MB
  bf16_t* x1_bf  = (bf16_t*)(ws + 88 * MB);   // 8 MB
  bf16_t* h_bf   = (bf16_t*)(ws + 24 * MB);   // 32 MB, reuse q/k/v/att region
  float*  ffo_f  = (float*) (ws + 56 * MB);   // reuse proj_f

  // 1) bf16 conversions
  k_to_bf16<<<2048, 256, 0, stream>>>(wq_w,  wq_bf,  CC * CC);
  k_to_bf16<<<2048, 256, 0, stream>>>(wk_w,  wk_bf,  CC * CC);
  k_to_bf16<<<2048, 256, 0, stream>>>(wv_w,  wv_bf,  CC * CC);
  k_to_bf16<<<2048, 256, 0, stream>>>(wo_w,  wo_bf,  CC * CC);
  k_to_bf16<<<2048, 256, 0, stream>>>(ff1_w, ff1_bf, FF * CC);
  k_to_bf16<<<2048, 256, 0, stream>>>(ff2_w, ff2_bf, CC * FF);
  k_to_bf16<<<2048, 256, 0, stream>>>(x,     x_bf,   MT * CC);

  // 2) QKV projections (bf16 out)
  dim3 gQKV(MT / BLK_M, CC / BLK_N);
  k_gemm_bf16<<<gQKV, 256, 0, stream>>>(x_bf, wq_bf, wq_b, nullptr, q_bf, CC, CC, 0);
  k_gemm_bf16<<<gQKV, 256, 0, stream>>>(x_bf, wk_bf, wk_b, nullptr, k_bf, CC, CC, 0);
  k_gemm_bf16<<<gQKV, 256, 0, stream>>>(x_bf, wv_bf, wv_b, nullptr, v_bf, CC, CC, 0);

  // 3) causal flash attention
  k_attention<<<dim3(TT / 64, 2 * NH), 128, 0, stream>>>(q_bf, k_bf, v_bf, att_bf);

  // 4) output projection (fp32 out)
  k_gemm_bf16<<<gQKV, 256, 0, stream>>>(att_bf, wo_bf, wo_b, proj_f, nullptr, CC, CC, 0);

  // 5) x1 = LN(x + proj)
  k_add_ln<<<MT, 256, 0, stream>>>(proj_f, x, ln1_g, ln1_b, x1_f, x1_bf);

  // 6) FFN: h = gelu(x1 @ ff1^T + b1) ; f = h @ ff2^T + b2
  k_gemm_bf16<<<dim3(MT / BLK_M, FF / BLK_N), 256, 0, stream>>>(
      x1_bf, ff1_bf, ff1_b, nullptr, h_bf, FF, CC, 1);
  k_gemm_bf16<<<dim3(MT / BLK_M, CC / BLK_N), 256, 0, stream>>>(
      h_bf, ff2_bf, ff2_b, ffo_f, nullptr, CC, FF, 0);

  // 7) out = LN(x1 + f)
  k_add_ln<<<MT, 256, 0, stream>>>(ffo_f, x1_f, ln2_g, ln2_b, (float*)d_out, nullptr);
}
